// IoUNet_7172595384502
// MI455X (gfx1250) — matrix operand order repaired and checked by reading, not compiled, not run on hardware
//
#include <hip/hip_runtime.h>

// ---------------------------------------------------------------------------
// IoU-Net head for MI455X (gfx1250): bf16 WMMA implicit-GEMM convs + WMMA FC.
// Triple-buffered GLOBAL_LOAD_ASYNC_TO_LDS pipeline with partial ASYNCcnt
// waits; 128x128 workgroup tiles, 32x64 wave tiles (8 WMMA per K-chunk).
// ---------------------------------------------------------------------------

#define DIM     256
#define HW      36
#define HWP     38             // padded
#define NIMG    32             // NI*NS
#define PTOT    (NIMG*HW*HW)   // 41472 spatial positions
#define KCONV   (9*DIM)        // 2304
#define NROI    512
#define KFC     (DIM*16)       // 4096
#define EPS_BN  1e-5f

typedef __attribute__((ext_vector_type(16))) __bf16 v16bf;
typedef __attribute__((ext_vector_type(8)))  float  v8f;

__device__ __forceinline__ __bf16 to_bf16(float f) {
    unsigned u = __builtin_bit_cast(unsigned, f);
    u += 0x7FFFu + ((u >> 16) & 1u);              // round-to-nearest-even
    unsigned short h = (unsigned short)(u >> 16);
    return __builtin_bit_cast(__bf16, h);
}

union FragB16 { v16bf v; uint4 q[2]; };

// low 32 bits of a flat shared-memory pointer == LDS byte offset (aperture map)
__device__ __forceinline__ unsigned lds_addr(const void* p) {
    return (unsigned)(unsigned long long)p;
}

// async L2->LDS copy, 16B per lane; INST_OFFSET shifts both src and dst
__device__ __forceinline__ void async_b128(unsigned lds, unsigned long long gaddr) {
    asm volatile("global_load_async_to_lds_b128 %0, %1, off" :: "v"(lds), "v"(gaddr) : "memory");
}
__device__ __forceinline__ void async_b128_o16(unsigned lds, unsigned long long gaddr) {
    asm volatile("global_load_async_to_lds_b128 %0, %1, off offset:16" :: "v"(lds), "v"(gaddr) : "memory");
}
__device__ __forceinline__ void async_b128_o32(unsigned lds, unsigned long long gaddr) {
    asm volatile("global_load_async_to_lds_b128 %0, %1, off offset:32" :: "v"(lds), "v"(gaddr) : "memory");
}
__device__ __forceinline__ void async_b128_o48(unsigned lds, unsigned long long gaddr) {
    asm volatile("global_load_async_to_lds_b128 %0, %1, off offset:48" :: "v"(lds), "v"(gaddr) : "memory");
}
__device__ __forceinline__ void wait_async0() {
    asm volatile("s_wait_asynccnt 0" ::: "memory");
}
// allow one younger 8-op stage group to remain in flight (in-order completion)
__device__ __forceinline__ void wait_async8() {
    asm volatile("s_wait_asynccnt 8" ::: "memory");
}

// ---------------------------------------------------------------------------
// Weight repack: [O][C][3][3] fp32  ->  bf16 [O][k], k = (r*3+s)*256 + c
// ---------------------------------------------------------------------------
__global__ void repack_w(const float* __restrict__ w, __bf16* __restrict__ wb) {
    int idx = blockIdx.x * 256 + threadIdx.x;      // 0 .. 256*2304-1
    int o  = idx / KCONV;
    int k  = idx - o * KCONV;
    int rs = k >> 8;
    int c  = k & 255;
    wb[idx] = to_bf16(w[(size_t)o * KCONV + c * 9 + rs]);
}

__global__ void cvt_bf16(const float* __restrict__ in, __bf16* __restrict__ out, int n) {
    int i = blockIdx.x * 256 + threadIdx.x;
    if (i < n) out[i] = to_bf16(in[i]);
}

// NCHW fp32 -> padded NHWC bf16 (halo pre-zeroed by memset)
__global__ void pack_input(const float* __restrict__ feat, __bf16* __restrict__ xp) {
    int c = threadIdx.x;
    int p = blockIdx.x;                 // 0..PTOT-1
    int n = p / (HW * HW);
    int rem = p - n * HW * HW;
    int yy = rem / HW, xx = rem - yy * HW;
    float v = feat[(((size_t)n * DIM + c) * HW + yy) * HW + xx];
    xp[(((size_t)n * HWP + yy + 1) * HWP + (xx + 1)) * DIM + c] = to_bf16(v);
}

// ---------------------------------------------------------------------------
// Conv as implicit GEMM:  Y[p][o] = sum_k Wb[o][k] * Xpad[p@(r,s)][c]
// WG tile 128(M=o) x 128(N=p); K staged 64/stage, 36 stages, triple-buffered.
// Wave tile 32x64 -> 16 WMMA per stage per wave.
// ---------------------------------------------------------------------------
__global__ __launch_bounds__(256) void conv_wmma(
    const __bf16* __restrict__ Wb,     // [256][2304]
    const __bf16* __restrict__ Xpad,   // [32][38][38][256]
    float* __restrict__ Y)             // [PTOT][256] (NHWC fp32)
{
    __shared__ __bf16 As[3][128][72];  // 144B rows (16B aligned), K=64 + pad
    __shared__ __bf16 Bs[3][128][72];

    const int tid  = threadIdx.x;
    const int wave = tid >> 5;
    const int lane = tid & 31;
    const int p0 = blockIdx.x * 128;
    const int o0 = blockIdx.y * 128;
    const int mbase = (wave & 3) * 32;   // 4 M-groups of 32
    const int nbase = (wave >> 2) * 64;  // 2 N-groups of 64

    // cooperative async-load mapping: per thread 64B of A and 64B of B / stage
    const int row = tid >> 1;            // 0..127
    const int col = (tid & 1) * 32;      // halfword col offset (0 or 32)

    int p = p0 + row;
    int nimg = p / (HW * HW);
    int rem  = p - nimg * HW * HW;
    int yy = rem / HW, xx = rem - yy * HW;
    const __bf16* bpos = Xpad + (((size_t)nimg * HWP + yy) * HWP + xx) * DIM + col;
    const __bf16* arow = Wb + (size_t)(o0 + row) * KCONV + col;

    // fragment addressing (ISA VGPR layouts)
    const int arow0 = mbase + (lane & 15);
    const int aklo  = (lane >> 4) * 8;    // A: split pattern
    const int bcol0 = nbase + (lane & 15);
    const int bklo  = (lane >> 4) * 8;    // B: split pattern

    v8f acc[2][4] = {};

    auto issue = [&](int st, int buf) {
        const int rs = st >> 2;                 // tap index 0..8
        const int c0 = (st & 3) << 6;           // channel base 0/64/128/192
        const int rr = rs / 3, ss = rs - rr * 3;
        unsigned da = lds_addr(&As[buf][row][col]);
        unsigned long long ga = (unsigned long long)(arow + rs * DIM + c0);
        async_b128(da, ga);  async_b128_o16(da, ga);
        async_b128_o32(da, ga);  async_b128_o48(da, ga);
        unsigned db = lds_addr(&Bs[buf][row][col]);
        unsigned long long gb = (unsigned long long)(bpos + (rr * HWP + ss) * DIM + c0);
        async_b128(db, gb);  async_b128_o16(db, gb);
        async_b128_o32(db, gb);  async_b128_o48(db, gb);
    };

    auto compute = [&](int buf) {
#pragma unroll
        for (int q = 0; q < 2; ++q) {           // two K=32 chunks
            const int cb = q * 32;
            FragB16 af[2], bf[4];
#pragma unroll
            for (int i = 0; i < 2; ++i) {
                af[i].q[0] = *(const uint4*)&As[buf][arow0 + 16 * i][cb + aklo];
                af[i].q[1] = *(const uint4*)&As[buf][arow0 + 16 * i][cb + aklo + 16];
            }
#pragma unroll
            for (int j = 0; j < 4; ++j) {
                bf[j].q[0] = *(const uint4*)&Bs[buf][bcol0 + 16 * j][cb + bklo];
                bf[j].q[1] = *(const uint4*)&Bs[buf][bcol0 + 16 * j][cb + bklo + 16];
            }
#pragma unroll
            for (int i = 0; i < 2; ++i)
#pragma unroll
                for (int j = 0; j < 4; ++j)
                    acc[i][j] = __builtin_amdgcn_wmma_f32_16x16x32_bf16(
                        false, af[i].v, false, bf[j].v,
                        (short)0, acc[i][j], false, false);
        }
    };

    issue(0, 0);
    issue(1, 1);
    for (int st = 0; st < 36; ++st) {
        if (st + 1 < 36) wait_async8(); else wait_async0();  // stage st resident
        __syncthreads();                                     // all waves ready
        if (st + 2 < 36) issue(st + 2, (st + 2) % 3);        // prefetch ahead
        compute(st % 3);
    }

    // store: D lane holds col n = lane%16, rows e + 8*(lane/16) -> contiguous o
#pragma unroll
    for (int i = 0; i < 2; ++i)
#pragma unroll
        for (int j = 0; j < 4; ++j) {
            int cp = p0 + nbase + j * 16 + (lane & 15);
            int ob = o0 + mbase + i * 16 + (lane >> 4) * 8;
            float* yp = Y + (size_t)cp * DIM + ob;
            v8f a = acc[i][j];
#pragma unroll
            for (int e = 0; e < 8; ++e) yp[e] = a[e];
        }
}

// per-channel sum / sumsq over NHWC fp32 Y
__global__ void chan_stats(const float* __restrict__ Y, float* cs, float* cq) {
    int c = threadIdx.x;
    size_t p0 = (size_t)blockIdx.x * 256;
    float s = 0.f, q = 0.f;
    for (int i = 0; i < 256; ++i) {
        float v = Y[(p0 + i) * DIM + c];
        s += v; q += v * v;
    }
    atomicAdd(&cs[c], s);
    atomicAdd(&cq[c], q);
}

// BN + ReLU, repack to padded NHWC bf16 for next conv
__global__ void bnrelu_pack(const float* __restrict__ Y, const float* cs, const float* cq,
                            const float* g, const float* beta, __bf16* __restrict__ xp) {
    int c = threadIdx.x;
    int p = blockIdx.x;
    int n = p / (HW * HW);
    int rem = p - n * HW * HW;
    int yy = rem / HW, xx = rem - yy * HW;
    const float inv = 1.f / (float)PTOT;
    float m = cs[c] * inv;
    float v = cq[c] * inv - m * m;
    float s = g[c] * rsqrtf(v + EPS_BN);
    float val = (Y[(size_t)p * DIM + c] - m) * s + beta[c];
    val = fmaxf(val, 0.f);
    xp[(((size_t)n * HWP + yy + 1) * HWP + (xx + 1)) * DIM + c] = to_bf16(val);
}

// BN + ReLU, write NCHW fp32 (input to PrRoI integral images)
__global__ void bnrelu_nchw(const float* __restrict__ Y, const float* cs, const float* cq,
                            const float* g, const float* beta, float* __restrict__ F) {
    int c = threadIdx.x;
    int p = blockIdx.x;
    int n = p / (HW * HW);
    int rem = p - n * HW * HW;
    int yy = rem / HW, xx = rem - yy * HW;
    const float inv = 1.f / (float)PTOT;
    float m = cs[c] * inv;
    float v = cq[c] * inv - m * m;
    float s = g[c] * rsqrtf(v + EPS_BN);
    float val = (Y[(size_t)p * DIM + c] - m) * s + beta[c];
    F[(((size_t)n * DIM + c) * HW + yy) * HW + xx] = fmaxf(val, 0.f);
}

// trapezoid cumsum along w (axis=0) or h (axis=1) of [32][256][36][36]
__global__ void trapcum(const float* __restrict__ in, float* __restrict__ out, int axis) {
    int line = blockIdx.x * 256 + threadIdx.x;
    if (line >= NIMG * DIM * HW) return;
    size_t base; int step;
    if (axis == 0) { base = (size_t)line * HW; step = 1; }
    else { int nc = line / HW; int w = line - nc * HW; base = (size_t)nc * HW * HW + w; step = HW; }
    float prev = in[base];
    out[base] = 0.f;
    float cum = 0.f;
    for (int i = 1; i < HW; ++i) {
        float cur = in[base + (size_t)i * step];
        cum += 0.5f * (prev + cur);
        out[base + (size_t)i * step] = cum;
        prev = cur;
    }
}

// PrRoI pooling: block = roi (512), thread = channel (256) -> rf bf16 [512][4096]
__global__ void prroi_pool(const float* __restrict__ F,  const float* __restrict__ Ux,
                           const float* __restrict__ Uy, const float* __restrict__ T,
                           const float* __restrict__ props, __bf16* __restrict__ rf) {
    int roi = blockIdx.x;
    int c   = threadIdx.x;
    int b   = roi >> 4;                          // 16 proposals per image
    const float* pr = props + (size_t)roi * 4;
    float x1 = pr[0] * 20.f, y1 = pr[1] * 20.f;
    float x2 = (pr[0] + pr[2]) * 20.f, y2 = (pr[1] + pr[3]) * 20.f;
    float bw = (x2 - x1) * 0.25f, bh = (y2 - y1) * 0.25f;

    float px0[5], px1[5], py0[5], py1[5];
    int wx[5], hy[5];
#pragma unroll
    for (int gi = 0; gi < 5; ++gi) {
        float xg = fminf(fmaxf(x1 + bw * (float)gi, 0.f), 35.f);
        float yg = fminf(fmaxf(y1 + bh * (float)gi, 0.f), 35.f);
        int wi = min(max((int)floorf(xg), 0), 34);
        int hi = min(max((int)floorf(yg), 0), 34);
        float sx = xg - (float)wi, sy = yg - (float)hi;
        px1[gi] = 0.5f * sx * sx; px0[gi] = sx - px1[gi];
        py1[gi] = 0.5f * sy * sy; py0[gi] = sy - py1[gi];
        wx[gi] = wi; hy[gi] = hi;
    }

    const size_t base = ((size_t)b * DIM + c) * (HW * HW);
    float G[5][5];
#pragma unroll
    for (int gy = 0; gy < 5; ++gy) {
        int h0 = hy[gy], h1 = h0 + 1;
#pragma unroll
        for (int gx = 0; gx < 5; ++gx) {
            int w0 = wx[gx], w1 = w0 + 1;
            size_t i00 = base + h0 * HW + w0;
            size_t i01 = base + h0 * HW + w1;
            size_t i10 = base + h1 * HW + w0;
            size_t i11 = base + h1 * HW + w1;
            G[gy][gx] = T[i00] + Uy[i00] * px0[gx] + Uy[i01] * px1[gx]
                + py0[gy] * (Ux[i00] + F[i00] * px0[gx] + F[i01] * px1[gx])
                + py1[gy] * (Ux[i10] + F[i10] * px0[gx] + F[i11] * px1[gx]);
        }
    }
    float area = bw * bh;
    float inv  = 1.f / fmaxf(area, 1e-8f);
    bool  ok   = area > 1e-8f;
#pragma unroll
    for (int py = 0; py < 4; ++py)
#pragma unroll
        for (int px = 0; px < 4; ++px) {
            float d = G[py + 1][px + 1] - G[py][px + 1] - G[py + 1][px] + G[py][px];
            float o = ok ? d * inv : 0.f;
            rf[(size_t)roi * KFC + c * 16 + py * 4 + px] = to_bf16(o);
        }
}

// ---------------------------------------------------------------------------
// FC GEMM:  fcy[r][o] = sum_k fcw[o][k] * rf[r][k]   (M=256, N=512, K=4096)
// Same triple-buffered async-LDS WMMA pipeline, 64 stages of K=64.
// ---------------------------------------------------------------------------
__global__ __launch_bounds__(256) void fc_wmma(
    const __bf16* __restrict__ Wb,   // [256][4096] bf16
    const __bf16* __restrict__ rf,   // [512][4096] bf16
    float* __restrict__ fcy)         // [512][256] fp32
{
    __shared__ __bf16 As[3][128][72];
    __shared__ __bf16 Bs[3][128][72];

    const int tid  = threadIdx.x;
    const int wave = tid >> 5;
    const int lane = tid & 31;
    const int p0 = blockIdx.x * 128;     // roi tile
    const int o0 = blockIdx.y * 128;     // out-feature tile
    const int mbase = (wave & 3) * 32;
    const int nbase = (wave >> 2) * 64;

    const int row = tid >> 1;
    const int col = (tid & 1) * 32;

    const __bf16* arow = Wb + (size_t)(o0 + row) * KFC + col;
    const __bf16* brow = rf + (size_t)(p0 + row) * KFC + col;

    const int arow0 = mbase + (lane & 15);
    const int aklo  = (lane >> 4) * 8;
    const int bcol0 = nbase + (lane & 15);
    const int bklo  = (lane >> 4) * 8;

    v8f acc[2][4] = {};

    auto issue = [&](int st, int buf) {
        const int kb = st << 6;
        unsigned da = lds_addr(&As[buf][row][col]);
        unsigned long long ga = (unsigned long long)(arow + kb);
        async_b128(da, ga);  async_b128_o16(da, ga);
        async_b128_o32(da, ga);  async_b128_o48(da, ga);
        unsigned db = lds_addr(&Bs[buf][row][col]);
        unsigned long long gb = (unsigned long long)(brow + kb);
        async_b128(db, gb);  async_b128_o16(db, gb);
        async_b128_o32(db, gb);  async_b128_o48(db, gb);
    };

    auto compute = [&](int buf) {
#pragma unroll
        for (int q = 0; q < 2; ++q) {
            const int cb = q * 32;
            FragB16 af[2], bf[4];
#pragma unroll
            for (int i = 0; i < 2; ++i) {
                af[i].q[0] = *(const uint4*)&As[buf][arow0 + 16 * i][cb + aklo];
                af[i].q[1] = *(const uint4*)&As[buf][arow0 + 16 * i][cb + aklo + 16];
            }
#pragma unroll
            for (int j = 0; j < 4; ++j) {
                bf[j].q[0] = *(const uint4*)&Bs[buf][bcol0 + 16 * j][cb + bklo];
                bf[j].q[1] = *(const uint4*)&Bs[buf][bcol0 + 16 * j][cb + bklo + 16];
            }
#pragma unroll
            for (int i = 0; i < 2; ++i)
#pragma unroll
                for (int j = 0; j < 4; ++j)
                    acc[i][j] = __builtin_amdgcn_wmma_f32_16x16x32_bf16(
                        false, af[i].v, false, bf[j].v,
                        (short)0, acc[i][j], false, false);
        }
    };

    issue(0, 0);
    issue(1, 1);
    for (int st = 0; st < 64; ++st) {
        if (st + 1 < 64) wait_async8(); else wait_async0();
        __syncthreads();
        if (st + 2 < 64) issue(st + 2, (st + 2) % 3);
        compute(st % 3);
    }

#pragma unroll
    for (int i = 0; i < 2; ++i)
#pragma unroll
        for (int j = 0; j < 4; ++j) {
            int cr = p0 + nbase + j * 16 + (lane & 15);      // roi
            int ob = o0 + mbase + i * 16 + (lane >> 4) * 8;  // out feature
            float* yp = fcy + (size_t)cr * DIM + ob;
            v8f a = acc[i][j];
#pragma unroll
            for (int e = 0; e < 8; ++e) yp[e] = a[e];
        }
}

__global__ void fc_stats(const float* __restrict__ fcy, float* fs, float* fq) {
    int o = threadIdx.x;
    float s = 0.f, q = 0.f;
    for (int r = 0; r < NROI; ++r) {
        float v = fcy[(size_t)r * DIM + o];
        s += v; q += v * v;
    }
    fs[o] = s; fq[o] = q;
}

// fused BN + ReLU + IoU head: out[r] = sum_o relu(bn(fcy[r][o])) * iou_w[o] + iou_b
__global__ void head(const float* __restrict__ fcy, const float* fs, const float* fq,
                     const float* g, const float* beta,
                     const float* iw, const float* ib, float* __restrict__ out) {
    __shared__ float sc[DIM], sh[DIM];
    int t = threadIdx.x;                 // 0..511 (roi)
    if (t < DIM) {
        float m = fs[t] * (1.f / (float)NROI);
        float v = fq[t] * (1.f / (float)NROI) - m * m;
        float s = g[t] * rsqrtf(v + EPS_BN);
        sc[t] = s; sh[t] = beta[t] - m * s;
    }
    __syncthreads();
    float acc = 0.f;
    const float* row = fcy + (size_t)t * DIM;
    for (int o = 0; o < DIM; ++o) {
        float val = fmaxf(row[o] * sc[o] + sh[o], 0.f);
        acc += val * iw[o];
    }
    out[t] = acc + ib[0];
}

// ---------------------------------------------------------------------------
extern "C" void kernel_launch(void* const* d_in, const int* in_sizes, int n_in,
                              void* d_out, int out_size, void* d_ws, size_t ws_size,
                              hipStream_t stream) {
    (void)in_sizes; (void)n_in; (void)out_size; (void)ws_size;

    const float* feat      = (const float*)d_in[0];
    const float* proposals = (const float*)d_in[1];
    const float* conv_w[3] = { (const float*)d_in[2], (const float*)d_in[6], (const float*)d_in[10] };
    const float* bn_g[3]   = { (const float*)d_in[4], (const float*)d_in[8], (const float*)d_in[12] };
    const float* bn_b[3]   = { (const float*)d_in[5], (const float*)d_in[9], (const float*)d_in[13] };
    const float* fc_w      = (const float*)d_in[14];
    const float* fcbn_g    = (const float*)d_in[16];
    const float* fcbn_b    = (const float*)d_in[17];
    const float* iou_w     = (const float*)d_in[18];
    const float* iou_b     = (const float*)d_in[19];
    float* out             = (float*)d_out;

    char* ws = (char*)d_ws;
    auto take = [&](size_t bytes) -> char* {
        char* p = ws;
        ws += (bytes + 255) & ~(size_t)255;
        return p;
    };

    const size_t XPAD_B = (size_t)NIMG * HWP * HWP * DIM * 2;
    const size_t FMAP_B = (size_t)PTOT * DIM * 4;

    __bf16* XA  = (__bf16*)take(XPAD_B);
    __bf16* XB  = (__bf16*)take(XPAD_B);
    float*  Y   = (float*) take(FMAP_B);
    __bf16* Wb[3];
    for (int l = 0; l < 3; ++l) Wb[l] = (__bf16*)take((size_t)DIM * KCONV * 2);
    __bf16* FCW = (__bf16*)take((size_t)DIM * KFC * 2);
    float*  F   = (float*) take(FMAP_B);
    float*  Ux  = (float*) take(FMAP_B);
    float*  Uy  = (float*) take(FMAP_B);
    float*  T   = (float*) take(FMAP_B);
    __bf16* RF  = (__bf16*)take((size_t)NROI * KFC * 2);
    float*  FCY = (float*) take((size_t)NROI * DIM * 4);
    float*  CS  = (float*) take(1024);
    float*  CQ  = (float*) take(1024);
    float*  FS  = (float*) take(1024);
    float*  FQ  = (float*) take(1024);

    // zero padded activation buffers (halo must be 0 every call)
    hipMemsetAsync(XA, 0, XPAD_B, stream);
    hipMemsetAsync(XB, 0, XPAD_B, stream);

    // weight conversions
    for (int l = 0; l < 3; ++l)
        repack_w<<<KCONV, 256, 0, stream>>>(conv_w[l], Wb[l]);
    cvt_bf16<<<(DIM * KFC + 255) / 256, 256, 0, stream>>>(fc_w, FCW, DIM * KFC);

    // input -> padded NHWC bf16
    pack_input<<<PTOT, 256, 0, stream>>>(feat, XA);

    __bf16* xin = XA;
    __bf16* xalt = XB;
    for (int l = 0; l < 3; ++l) {
        dim3 grid(PTOT / 128, DIM / 128);
        conv_wmma<<<grid, 256, 0, stream>>>(Wb[l], xin, Y);
        hipMemsetAsync(CS, 0, DIM * 4, stream);
        hipMemsetAsync(CQ, 0, DIM * 4, stream);
        chan_stats<<<PTOT / 256, 256, 0, stream>>>(Y, CS, CQ);
        if (l < 2) {
            bnrelu_pack<<<PTOT, 256, 0, stream>>>(Y, CS, CQ, bn_g[l], bn_b[l], xalt);
            __bf16* t = xin; xin = xalt; xalt = t;
        } else {
            bnrelu_nchw<<<PTOT, 256, 0, stream>>>(Y, CS, CQ, bn_g[2], bn_b[2], F);
        }
    }

    // integral images
    const int NLINES = NIMG * DIM * HW;
    trapcum<<<(NLINES + 255) / 256, 256, 0, stream>>>(F, Ux, 0);
    trapcum<<<(NLINES + 255) / 256, 256, 0, stream>>>(F, Uy, 1);
    trapcum<<<(NLINES + 255) / 256, 256, 0, stream>>>(Ux, T, 1);

    prroi_pool<<<NROI, 256, 0, stream>>>(F, Ux, Uy, T, proposals, RF);

    fc_wmma<<<dim3(NROI / 128, DIM / 128), 256, 0, stream>>>(FCW, RF, FCY);
    fc_stats<<<1, 256, 0, stream>>>(FCY, FS, FQ);
    head<<<1, NROI, 0, stream>>>(FCY, FS, FQ, fcbn_g, fcbn_b, iou_w, iou_b, out);
}
